// CrossScaleAggregationModule_3547642986607
// MI455X (gfx1250) — matrix operand order repaired and checked by reading, not compiled, not run on hardware
//
#include <hip/hip_runtime.h>
#include <hip/hip_bf16.h>

#define NCAND 65536
#define NPTS  1024
#define DD    256
#define CC    256
#define SLABS 8
#define NEGF  (-3.0e38f)

typedef __attribute__((ext_vector_type(16))) __bf16 v16bf;
typedef __attribute__((ext_vector_type(8)))  __bf16 v8bf;
typedef __attribute__((ext_vector_type(8)))  float  v8f;
typedef __attribute__((ext_vector_type(4)))  unsigned int u32x4;
typedef __attribute__((ext_vector_type(8)))  int i32x8;
typedef __attribute__((ext_vector_type(4)))  int i32x4;

__device__ __forceinline__ unsigned short f2bf(float x) {
    __bf16 h = (__bf16)x;
    unsigned short u;
    __builtin_memcpy(&u, &h, 2);
    return u;
}
__device__ __forceinline__ __bf16 bits2bf(unsigned short u) {
    __bf16 h;
    __builtin_memcpy(&h, &u, 2);
    return h;
}

// ---------------------------------------------------------------------------
// TDM descriptor helpers (CDNA5 Tensor Data Mover, D# per ISA 08_async_tensor)
// group0: [1:0]=count, [63:32]=lds_addr, [120:64]=global_addr, [127:126]=type=2
// group1: [17:16]=data_size, [79:48]=tensor_dim0, [111:80]=tensor_dim1,
//         [127:112]=tile_dim0, [143:128]=tile_dim1, [207:160]=tensor_dim0_stride
// amdgpu-toolchain form: 6 args (g0, g1, g2, g3, g4, cpol)
// ---------------------------------------------------------------------------
__device__ __forceinline__ void tdm_load_1d_b32(unsigned int lds_off, const void* g,
                                                unsigned int nwords) {
    unsigned long long ga = (unsigned long long)(size_t)g;
    u32x4 g0;
    g0[0] = 1u;                                            // count=1
    g0[1] = lds_off;
    g0[2] = (unsigned int)(ga & 0xFFFFFFFFu);
    g0[3] = (unsigned int)((ga >> 32) & 0x01FFFFFFu) | (2u << 30);   // type=2
    i32x8 g1;
    g1[0] = (int)(2u << 16);                               // data_size = 4B
    g1[1] = (int)((nwords & 0xFFFFu) << 16);               // tensor_dim0[15:0]
    g1[2] = (int)(((nwords >> 16) & 0xFFFFu) | (1u << 16));// tensor_dim0[31:16], tensor_dim1=1
    g1[3] = (int)((nwords & 0xFFFFu) << 16);               // tile_dim0
    g1[4] = 0;                                             // tile_dim1=0 (1-D)
    g1[5] = (int)nwords;                                   // tensor_dim0_stride lo
    g1[6] = 0;
    g1[7] = 0;
    i32x4 z4 = {0, 0, 0, 0};
    i32x8 z8 = {0, 0, 0, 0, 0, 0, 0, 0};
    __builtin_amdgcn_tensor_load_to_lds(g0, g1, z4, z4, z8, 0);
}

__device__ __forceinline__ void tdm_load_2d_b16(unsigned int lds_off, const void* g,
                                                unsigned int tile_x, unsigned int tile_y,
                                                unsigned int stride_elems) {
    unsigned long long ga = (unsigned long long)(size_t)g;
    u32x4 g0;
    g0[0] = 1u;
    g0[1] = lds_off;
    g0[2] = (unsigned int)(ga & 0xFFFFFFFFu);
    g0[3] = (unsigned int)((ga >> 32) & 0x01FFFFFFu) | (2u << 30);
    i32x8 g1;
    g1[0] = (int)(1u << 16);                               // data_size = 2B
    g1[1] = (int)((stride_elems & 0xFFFFu) << 16);         // tensor_dim0[15:0]
    g1[2] = (int)(((stride_elems >> 16) & 0xFFFFu) |       // tensor_dim0[31:16]
                  ((tile_y & 0xFFFFu) << 16));             // tensor_dim1[15:0]
    g1[3] = (int)((tile_x & 0xFFFFu) << 16);               // tile_dim0 (dim1 hi = 0)
    g1[4] = (int)(tile_y & 0xFFFFu);                       // tile_dim1
    g1[5] = (int)stride_elems;                             // tensor_dim0_stride lo
    g1[6] = 0;
    g1[7] = 0;
    i32x4 z4 = {0, 0, 0, 0};
    i32x8 z8 = {0, 0, 0, 0, 0, 0, 0, 0};
    __builtin_amdgcn_tensor_load_to_lds(g0, g1, z4, z4, z8, 0);
}

// ---------------------------------------------------------------------------
// Kernel 0: pre-swizzle Wk into exact 32x16 bf16 B-fragment layout:
// wkb[((k*16+ct)*32 + lane)*16 + e] = Wk[k*32 + e + 16*(lane>=16)][ct*16 + lane%16]
// ---------------------------------------------------------------------------
__global__ void __launch_bounds__(256)
csam_wkprep(const float* __restrict__ Wk, unsigned short* __restrict__ wkb)
{
    int gid = blockIdx.x * 256 + threadIdx.x;   // 8*16*32*16 = 65536
    int e    = gid & 15;
    int lane = (gid >> 4) & 31;
    int ct   = (gid >> 9) & 15;
    int k    = gid >> 13;
    int lh = lane >> 4, n16 = lane & 15;
    wkb[gid] = f2bf(Wk[(size_t)(k * 32 + e + 16 * lh) * CC + ct * 16 + n16]);
}

// ---------------------------------------------------------------------------
// Kernel 1: query = points_feat @ Wq + bq  -> bf16 row-major [NPTS][CC]
// ---------------------------------------------------------------------------
__global__ void __launch_bounds__(256)
csam_query(const float* __restrict__ points_feat, const float* __restrict__ Wq,
           const float* __restrict__ bq, unsigned short* __restrict__ qbf)
{
    int gid = blockIdx.x * 256 + threadIdx.x;
    int p = gid >> 8;
    int c = gid & 255;
    float s = bq[c];
    const float* pf = points_feat + (size_t)p * DD;
    for (int k = 0; k < DD; ++k)
        s += pf[k] * Wq[(size_t)k * CC + c];
    qbf[gid] = f2bf(s);
}

// ---------------------------------------------------------------------------
// Kernel 2: keyf = (box_feat * scales[lvl]) @ Wk + bk -> bf16 [NCAND][CC]
// Also emits: boxt (box_feat transposed bf16, [DD][NCAND]) and cand_xy.
// ---------------------------------------------------------------------------
__global__ void __launch_bounds__(256)
csam_keyf(const float* __restrict__ box_feat, const float* __restrict__ centers,
          const unsigned short* __restrict__ wkb, const float* __restrict__ bk,
          const float* __restrict__ scales, unsigned short* __restrict__ keyf,
          unsigned short* __restrict__ boxt, float* __restrict__ cand_xy)
{
    const int wv   = threadIdx.x >> 5;
    const int lane = threadIdx.x & 31;
    const int n16  = lane & 15;
    const int lh   = lane >> 4;
    const int cbase  = (blockIdx.x * 8 + wv) * 16;
    const int mycand = cbase + n16;

    float yy = centers[mycand * 4 + 0];
    float xx = centers[mycand * 4 + 1];
    float st = centers[mycand * 4 + 2];
    float half = floorf(st * 0.5f);
    float cy = yy + half, cx = xx + half;
    int lvl = 31 - __clz((int)(st + 0.5f)) - 3;   // st = 2^(lvl+3)
    lvl = lvl < 0 ? 0 : (lvl > 4 ? 4 : lvl);
    float wsc = scales[lvl];
    if (lh == 0) {
        cand_xy[mycand * 2 + 0] = cx;
        cand_xy[mycand * 2 + 1] = cy;
    }

    // A fragments (16x32 bf16): lane row M = lane%16; e -> K = e + 16*lh
    v16bf afr[8];
    const float* row = box_feat + (size_t)mycand * DD;
    #pragma unroll
    for (int k = 0; k < 8; ++k) {
        int b = k * 32 + 16 * lh;
        #pragma unroll
        for (int e = 0; e < 8; ++e) {
            afr[k][e]     = (__bf16)(row[b + e]      * wsc);
            afr[k][e + 8] = (__bf16)(row[b + 8 + e] * wsc);
        }
    }
    // transposed unscaled bf16 copy of box_feat (for merge-GEMM B fragments)
    if (lh == 0) {
        for (int d = 0; d < DD; ++d)
            boxt[(size_t)d * NCAND + mycand] = f2bf(row[d]);
    }
    for (int ct = 0; ct < 16; ++ct) {
        v8f c = {};
        #pragma unroll
        for (int k = 0; k < 8; ++k) {
            v16bf bfr = *(const v16bf*)(wkb + ((size_t)(k * 16 + ct) * 32 + lane) * 16);
            c = __builtin_amdgcn_wmma_f32_16x16x32_bf16(false, afr[k], false, bfr,
                                                        (short)0, c, false, false);
        }
        float bias = bk[ct * 16 + n16];
        #pragma unroll
        for (int r = 0; r < 8; ++r) {
            int cand = cbase + r + 8 * lh;
            keyf[(size_t)cand * CC + ct * 16 + n16] = f2bf(c[r] + bias);
        }
    }
}

// ---------------------------------------------------------------------------
// Kernel 3: fused sim + masked online softmax + merge, TDM double-buffered.
// Block = 8 waves x 16 points; all waves share each 32-candidate chunk staged
// once into LDS by wave 0 via tensor_load_to_lds (TENSORcnt + barrier).
// ---------------------------------------------------------------------------
__global__ void __launch_bounds__(256)
csam_flash(const float* __restrict__ boxes,
           const unsigned short* __restrict__ qbf,
           const unsigned short* __restrict__ keyf,
           const unsigned short* __restrict__ boxt,
           const float* __restrict__ cand_xy,
           float* __restrict__ pm, float* __restrict__ ps,
           float* __restrict__ pacc)
{
    __shared__ unsigned short s_keyf[2][32 * 256];   // [cand][c] bf16, 16 KB each
    __shared__ unsigned short s_boxt[2][256 * 32];   // [d][cand] bf16, 16 KB each
    __shared__ float          s_cxy[2][64];          // [cand][{x,y}]
    __shared__ unsigned short pbuf[8][16][32];       // [wave][point][cand] attn^T stage

    const int wv   = threadIdx.x >> 5;
    const int lane = threadIdx.x & 31;
    const int n16  = lane & 15;
    const int lh   = lane >> 4;
    const int pt_base = (blockIdx.x * 8 + wv) * 16;
    const int pt = pt_base + n16;
    const int slab = blockIdx.y;
    const int c_begin = slab * (NCAND / SLABS);
    const int NITER = (NCAND / SLABS) / 32;

    const float bx1 = boxes[pt * 4 + 0], by1 = boxes[pt * 4 + 1];
    const float bx2 = boxes[pt * 4 + 2], by2 = boxes[pt * 4 + 3];

    // Persistent q B-fragments: element e -> K = k*32 + e + 16*lh (contiguous)
    v16bf bq[8];
    #pragma unroll
    for (int k = 0; k < 8; ++k)
        bq[k] = *(const v16bf*)(qbf + (size_t)pt * CC + k * 32 + 16 * lh);

    v8f acc[16];
    #pragma unroll
    for (int t = 0; t < 16; ++t) acc[t] = (v8f){};

    float m = NEGF, ssum = 0.0f;

    // prologue: stage chunk 0 into buffer 0
    if (wv == 0) {
        tdm_load_1d_b32((unsigned int)(size_t)&s_keyf[0][0],
                        keyf + (size_t)c_begin * CC, (32 * 256 * 2) / 4);
        tdm_load_2d_b16((unsigned int)(size_t)&s_boxt[0][0],
                        boxt + c_begin, 32, 256, NCAND);
        tdm_load_1d_b32((unsigned int)(size_t)&s_cxy[0][0],
                        cand_xy + (size_t)c_begin * 2, 64);
        __builtin_amdgcn_s_wait_tensorcnt(0);
    }
    __syncthreads();

    for (int it = 0; it < NITER; ++it) {
        const int c0 = c_begin + it * 32;
        const int cur = it & 1, nxt = cur ^ 1;

        if (wv == 0 && it + 1 < NITER) {
            const int cn = c0 + 32;
            tdm_load_1d_b32((unsigned int)(size_t)&s_keyf[nxt][0],
                            keyf + (size_t)cn * CC, (32 * 256 * 2) / 4);
            tdm_load_2d_b16((unsigned int)(size_t)&s_boxt[nxt][0],
                            boxt + cn, 32, 256, NCAND);
            tdm_load_1d_b32((unsigned int)(size_t)&s_cxy[nxt][0],
                            cand_xy + (size_t)cn * 2, 64);
        }

        v8f sv[2];
        int mk[2];
        #pragma unroll
        for (int h = 0; h < 2; ++h) {
            const int crow = h * 16 + n16;             // A row within chunk
            v8f ctile = (v8f){};
            #pragma unroll
            for (int k = 0; k < 8; ++k) {
                const unsigned short* rp = &s_keyf[cur][crow * 256 + k * 32 + 16 * lh];
                v8bf lo = *(const v8bf*)rp;
                v8bf hi = *(const v8bf*)(rp + 8);
                v16bf a;
                #pragma unroll
                for (int e = 0; e < 8; ++e) { a[e] = lo[e]; a[e + 8] = hi[e]; }
                ctile = __builtin_amdgcn_wmma_f32_16x16x32_bf16(false, a, false, bq[k],
                                                                (short)0, ctile, false, false);
            }
            int mbits = 0;
            #pragma unroll
            for (int r = 0; r < 8; ++r) {
                int ci = h * 16 + r + 8 * lh;          // chunk-local candidate
                float cx  = s_cxy[cur][ci * 2 + 0];
                float cyv = s_cxy[cur][ci * 2 + 1];
                float l  = cx  - bx1, tt = cyv - by1;
                float rr = bx2 - cx,  bb = by2 - cyv;
                float mn = fminf(fminf(l, tt), fminf(rr, bb));
                float s  = fminf(fmaxf(ctile[r], -50.0f), 50.0f);
                bool msk = mn > 0.0f;
                ctile[r] = msk ? s : NEGF;
                mbits |= msk ? (1 << r) : 0;
            }
            sv[h] = ctile; mk[h] = mbits;
        }
        // per-point-column chunk max (lanes n and n+16 hold the same column)
        float tmax = NEGF;
        #pragma unroll
        for (int h = 0; h < 2; ++h)
            #pragma unroll
            for (int r = 0; r < 8; ++r) tmax = fmaxf(tmax, sv[h][r]);
        tmax = fmaxf(tmax, __shfl_xor(tmax, 16, 32));
        float mnew = fmaxf(m, tmax);
        float corr = __expf(m - mnew);

        float rsum = 0.0f;
        #pragma unroll
        for (int h = 0; h < 2; ++h) {
            #pragma unroll
            for (int r = 0; r < 8; ++r) {
                float p = ((mk[h] >> r) & 1) ? __expf(sv[h][r] - mnew) : 0.0f;
                rsum += p;
                pbuf[wv][n16][h * 16 + r + 8 * lh] = f2bf(p);
            }
        }
        rsum += __shfl_xor(rsum, 16, 32);
        ssum = ssum * corr + rsum;
        m = mnew;

        // rescale accumulator rows (point r+8*lh) via ds_bpermute broadcast
        #pragma unroll
        for (int r = 0; r < 8; ++r) {
            float cr = __shfl(corr, r + 8 * lh, 32);
            #pragma unroll
            for (int t = 0; t < 16; ++t) acc[t][r] *= cr;
        }

        asm volatile("s_wait_dscnt 0" ::: "memory");   // intra-wave LDS transpose RAW

        // A fragment of attn^T: row M = point (lane%16), K = e + 16*lh
        v16bf ap;
        {
            const unsigned short* rowp = &pbuf[wv][n16][0];
        int off = 16 * lh;
            #pragma unroll
            for (int e = 0; e < 8; ++e) {
                ap[e]     = bits2bf(rowp[off + e]);
            ap[e + 8] = bits2bf(rowp[off + 8 + e]);
            }
        }
        // merge: acc[t] += attn^T (16x32) @ box_feat chunk (32x16 per d-tile);
        // B fragment is one contiguous 32B LDS read from the transposed chunk.
        #pragma unroll
        for (int t = 0; t < 16; ++t) {
            v16bf bb_ = *(const v16bf*)&s_boxt[cur][(t * 16 + n16) * 32 + 16 * lh];
            acc[t] = __builtin_amdgcn_wmma_f32_16x16x32_bf16(false, ap, false, bb_,
                                                             (short)0, acc[t], false, false);
        }

        if (wv == 0) __builtin_amdgcn_s_wait_tensorcnt(0);
        __syncthreads();
    }

    if (lh == 0) {
        pm[slab * NPTS + pt] = m;
        ps[slab * NPTS + pt] = ssum;
    }
    #pragma unroll
    for (int t = 0; t < 16; ++t) {
        #pragma unroll
        for (int r = 0; r < 8; ++r) {
            int prow = pt_base + r + 8 * lh;
            pacc[((size_t)slab * NPTS + prow) * DD + t * 16 + n16] = acc[t][r];
        }
    }
}

// ---------------------------------------------------------------------------
// Kernel 4: log-sum-exp combine of slab partials + residual add
// ---------------------------------------------------------------------------
__global__ void __launch_bounds__(256)
csam_final(const float* __restrict__ points_feat,
           const float* __restrict__ pm, const float* __restrict__ ps,
           const float* __restrict__ pacc, float* __restrict__ out)
{
    int p = blockIdx.x;
    int d = threadIdx.x;
    float M = NEGF;
    #pragma unroll
    for (int k = 0; k < SLABS; ++k) M = fmaxf(M, pm[k * NPTS + p]);
    float S = 0.0f, V = 0.0f;
    #pragma unroll
    for (int k = 0; k < SLABS; ++k) {
        float w = __expf(pm[k * NPTS + p] - M);
        S += w * ps[k * NPTS + p];
        V += w * pacc[((size_t)k * NPTS + p) * DD + d];
    }
    float merge = (S > 0.0f) ? (V / S) : 0.0f;
    out[(size_t)p * DD + d] = points_feat[(size_t)p * DD + d] + merge;
}

// ---------------------------------------------------------------------------
extern "C" void kernel_launch(void* const* d_in, const int* in_sizes, int n_in,
                              void* d_out, int out_size, void* d_ws, size_t ws_size,
                              hipStream_t stream)
{
    (void)in_sizes; (void)n_in; (void)out_size; (void)ws_size;
    const float* points_feat = (const float*)d_in[0];
    const float* box_feat    = (const float*)d_in[1];
    const float* centers     = (const float*)d_in[2];
    const float* boxes       = (const float*)d_in[3];
    const float* Wq          = (const float*)d_in[4];
    const float* bq          = (const float*)d_in[5];
    const float* Wk          = (const float*)d_in[6];
    const float* bk          = (const float*)d_in[7];
    const float* scales      = (const float*)d_in[8];
    float* out = (float*)d_out;

    char* ws = (char*)d_ws;
    size_t o = 0;
    unsigned short* qbf  = (unsigned short*)(ws + o); o += (size_t)NPTS * CC * 2;   // 512 KB
    unsigned short* keyf = (unsigned short*)(ws + o); o += (size_t)NCAND * CC * 2;  // 32 MB
    unsigned short* boxt = (unsigned short*)(ws + o); o += (size_t)NCAND * DD * 2;  // 32 MB
    float* cand_xy = (float*)(ws + o); o += (size_t)NCAND * 2 * 4;                  // 512 KB
    unsigned short* wkb = (unsigned short*)(ws + o); o += (size_t)65536 * 2;        // 128 KB
    float* pm   = (float*)(ws + o); o += (size_t)SLABS * NPTS * 4;
    float* ps   = (float*)(ws + o); o += (size_t)SLABS * NPTS * 4;
    float* pacc = (float*)(ws + o); o += (size_t)SLABS * NPTS * DD * 4;             // 8 MB

    csam_wkprep<<<dim3(65536 / 256), dim3(256), 0, stream>>>(Wk, wkb);
    csam_query<<<dim3(NPTS * CC / 256), dim3(256), 0, stream>>>(points_feat, Wq, bq, qbf);
    csam_keyf<<<dim3(NCAND / 128), dim3(256), 0, stream>>>(box_feat, centers, wkb, bk,
                                                           scales, keyf, boxt, cand_xy);
    csam_flash<<<dim3(NPTS / 128, SLABS), dim3(256), 0, stream>>>(boxes, qbf, keyf, boxt,
                                                                  cand_xy, pm, ps, pacc);
    csam_final<<<dim3(NPTS), dim3(256), 0, stream>>>(points_feat, pm, ps, pacc, out);
}
